// CrossAttentionEinsum_57501022159094
// MI455X (gfx1250) — matrix operand order repaired and checked by reading, not compiled, not run on hardware
//
#include <hip/hip_runtime.h>
#include <stdint.h>

// ---------------------------------------------------------------------------
// CrossAttentionEinsum for MI455X (gfx1250), wave32 + WMMA bf16.
//
// Algebraic restructure: K/V are never materialized.
//   scores[h,m] = ctx[m,:] . (Wk_h^T q_h)           (R precomputed per pos)
//   out         = concat_h(attn_h^T @ ctx) @ (Wv_h @ Wo_h)   (Wcat precomputed)
// FLOPs drop ~16x; the binding constraint is one streaming read of context
// (256 MB -> ~11 us @ 23.3 TB/s). All weights stay resident in the 192 MB L2.
//
// v2: fragment-friendly LDS layouts so WMMA operands load as ds_load_b128
// pairs + bit_cast (no per-element u16 loads / v_mov_b16 merges).
// ---------------------------------------------------------------------------

typedef __bf16 bf16;
typedef __attribute__((ext_vector_type(16))) __bf16    v16bf;
typedef __attribute__((ext_vector_type(8)))  float     v8f;
typedef __attribute__((ext_vector_type(4)))  uint32_t  v4u;
typedef __attribute__((ext_vector_type(8)))  uint32_t  v8u;

#define NPOS   4096   // B*L
#define MCTX   64     // context tokens per position
#define DCC    256    // context dim
#define DQQ    512    // x dim == inner
#define NHEAD  8
#define DHEAD  64

static __device__ __forceinline__ v8f wmma_bf16(v16bf a, v16bf b, v8f c) {
  return __builtin_amdgcn_wmma_f32_16x16x32_bf16(false, a, false, b,
                                                 (short)0, c, false, false);
}

// 32 contiguous, 16B-aligned bytes -> v16bf (two b128 LDS loads, no merges)
static __device__ __forceinline__ v16bf frag16(const bf16* p) {
  v4u lo = *(const v4u*)p;
  v4u hi = *(const v4u*)(p + 8);
  v8u w  = __builtin_shufflevector(lo, hi, 0, 1, 2, 3, 4, 5, 6, 7);
  return __builtin_bit_cast(v16bf, w);
}

// A fragment (16M x 32K) from a row-major tile; ld multiple of 8 halfs,
// base 16B aligned. ISA 7.12.2: lane L = row L%16; elems 0..7 = K k0..k0+7,
// elems 8..15 = K k0+16..k0+23, k0=(L/16)*8  -> two aligned 16B chunks.
static __device__ __forceinline__ v16bf load_A_rm(const bf16* base, int ld) {
  int lane = threadIdx.x & 31;
  const bf16* p = base + (size_t)(lane & 15) * ld + (lane >> 4) * 8;
  v4u lo = *(const v4u*)p;
  v4u hi = *(const v4u*)(p + 16);
  v8u w  = __builtin_shufflevector(lo, hi, 0, 1, 2, 3, 4, 5, 6, 7);
  return __builtin_bit_cast(v16bf, w);
}

// B fragment (32K x 16N) from col-major [n][k] storage (ldn halfs, mult of 8):
// lane L holds col L%16, K = k0..k0+7, k0+16..k0+23, k0=(L/16)*8.
static __device__ __forceinline__ v16bf load_B_cm(const bf16* base, int ldn) {
  int lane = threadIdx.x & 31;
  const bf16* p = base + (size_t)(lane & 15) * ldn + (lane >> 4) * 8;
  v4u lo = *(const v4u*)p;
  v4u hi = *(const v4u*)(p + 16);
  v8u w  = __builtin_shufflevector(lo, hi, 0, 1, 2, 3, 4, 5, 6, 7);
  return __builtin_bit_cast(v16bf, w);
}

// B fragment from row-major [k][n] storage (scalar; only where a transpose
// is unavoidable: attn T-phase reading ctx).
static __device__ __forceinline__ v16bf load_B_rm_scalar(const bf16* base,
                                                         int ld) {
  int lane = threadIdx.x & 31;
  int n = lane & 15, h = lane >> 4;
  const bf16* p = base + (size_t)(h * 8) * ld + n;
  v16bf b;
#pragma unroll
  for (int j = 0; j < 16; ++j) b[j] = p[(size_t)(j < 8 ? j : j + 8) * ld];
  return b;
}

static __device__ __forceinline__ uint32_t pack2(float x, float y) {
  uint32_t a = (uint32_t)__builtin_bit_cast(uint16_t, (bf16)x);
  uint32_t b = (uint32_t)__builtin_bit_cast(uint16_t, (bf16)y);
  return a | (b << 16);
}

// ---------------------------------------------------------------------------
// Prep kernels
// ---------------------------------------------------------------------------
__global__ void k_f32_to_bf16(const float* __restrict__ src,
                              bf16* __restrict__ dst, int n) {
  int i = blockIdx.x * blockDim.x + threadIdx.x;
  if (i < n) dst[i] = (bf16)src[i];
}

// WkT[j][c] = Wk[c][j]   (512 x 256 bf16)
__global__ void k_build_wkt(const float* __restrict__ Wk,
                            bf16* __restrict__ WkT) {
  int i = blockIdx.x * blockDim.x + threadIdx.x;
  if (i >= DQQ * DCC) return;
  int j = i >> 8, c = i & 255;
  WkT[i] = (bf16)Wk[c * DQQ + j];
}

// Wcat[h*256+c][j] = sum_d Wv[c][h*64+d] * Wo[h*64+d][j]   (2048 x 512 bf16)
__global__ void k_build_wcat(const float* __restrict__ Wv,
                             const float* __restrict__ Wo,
                             bf16* __restrict__ Wcat) {
  int i = blockIdx.x * blockDim.x + threadIdx.x;
  if (i >= 2048 * 512) return;
  int row = i >> 9, j = i & 511;
  int h = row >> 8, c = row & 255;
  float acc = 0.f;
#pragma unroll 8
  for (int d = 0; d < DHEAD; ++d)
    acc += Wv[c * DQQ + h * DHEAD + d] * Wo[(h * DHEAD + d) * DQQ + j];
  Wcat[i] = (bf16)acc;
}

// ---------------------------------------------------------------------------
// Shared GEMM block: 256 thr (8 waves), 64x64 tile; A staged in fragment
// order (one b128 global load -> one b128 LDS store per thread), B staged
// col-major for vectorized B fragments.
// ---------------------------------------------------------------------------
#define BT_LD 40  // col-major B tile row stride (halfs); 80 B, 16B aligned

// Kernel: Qs = (x @ Wq) * DIM_HEAD^-0.5, bf16 out.  M=4096 N=512 K=512.
__global__ __launch_bounds__(256)
void k_gemm_q(const bf16* __restrict__ A, const bf16* __restrict__ B,
              bf16* __restrict__ D) {
  __shared__ bf16 AsF[4 * 32 * 16];   // [m-tile][lane][16 halfs] fragments
  __shared__ bf16 BsT[64 * BT_LD];    // col-major [n][k]
  int m0 = blockIdx.y * 64, n0 = blockIdx.x * 64;
  int t = threadIdx.x, wave = t >> 5, lane = t & 31;
  int mt = wave >> 1, ntb = (wave & 1) * 2;
  v8f acc0 = {}, acc1 = {};
  for (int k0 = 0; k0 < DQQ; k0 += 32) {
    { int idx = t * 8, r = idx >> 5, c = idx & 31;
      v4u val = *(const v4u*)(A + (size_t)(m0 + r) * DQQ + k0 + c);
      int amt = r >> 4, alane = (r & 15) + ((c >> 3) & 1) * 16;
      *(v4u*)&AsF[(amt * 32 + alane) * 16 + (c >> 4) * 8] = val; }
    { int idx = t * 8, r = idx >> 6, c = idx & 63;
      union { v4u v; bf16 h[8]; } u;
      u.v = *(const v4u*)(B + (size_t)(k0 + r) * DQQ + n0 + c);
#pragma unroll
      for (int j = 0; j < 8; ++j) BsT[(c + j) * BT_LD + r] = u.h[j]; }
    __syncthreads();
    v16bf a  = frag16(&AsF[(mt * 32 + lane) * 16]);
    v16bf b0 = load_B_cm(&BsT[(ntb * 16) * BT_LD], BT_LD);
    v16bf b1 = load_B_cm(&BsT[((ntb + 1) * 16) * BT_LD], BT_LD);
    acc0 = wmma_bf16(a, b0, acc0);
    acc1 = wmma_bf16(a, b1, acc1);
    __syncthreads();
  }
  const float scale = 0.125f;  // 64^-0.5 folded into Q
  int col = lane & 15, r0 = (lane >> 4) * 8;
#pragma unroll
  for (int i = 0; i < 8; ++i) {
    size_t row = m0 + mt * 16 + r0 + i;
    D[row * DQQ + n0 + ntb * 16 + col]       = (bf16)(acc0[i] * scale);
    D[row * DQQ + n0 + (ntb + 1) * 16 + col] = (bf16)(acc1[i] * scale);
  }
}

// Kernel: per-head R_h = Q_h @ Wk_h^T.  M=4096 N=256 K=64, grid.z = head.
// Output as R[pos][h][c] (head-major: vector-loadable in k_attn).
__global__ __launch_bounds__(256)
void k_gemm_r(const bf16* __restrict__ Qs, const bf16* __restrict__ WkT,
              bf16* __restrict__ R) {
  __shared__ bf16 AsF[4 * 32 * 16];
  __shared__ bf16 BsT[64 * BT_LD];
  int m0 = blockIdx.y * 64, n0 = blockIdx.x * 64, hh = blockIdx.z;
  int t = threadIdx.x, wave = t >> 5, lane = t & 31;
  int mt = wave >> 1, ntb = (wave & 1) * 2;
  v8f acc0 = {}, acc1 = {};
  for (int k0 = 0; k0 < DHEAD; k0 += 32) {
    { int idx = t * 8, r = idx >> 5, c = idx & 31;
      v4u val = *(const v4u*)(Qs + (size_t)(m0 + r) * DQQ + hh * DHEAD + k0 + c);
      int amt = r >> 4, alane = (r & 15) + ((c >> 3) & 1) * 16;
      *(v4u*)&AsF[(amt * 32 + alane) * 16 + (c >> 4) * 8] = val; }
    { int idx = t * 8, r = idx >> 6, c = idx & 63;
      union { v4u v; bf16 h[8]; } u;
      u.v = *(const v4u*)(WkT + (size_t)(hh * DHEAD + k0 + r) * DCC + n0 + c);
#pragma unroll
      for (int j = 0; j < 8; ++j) BsT[(c + j) * BT_LD + r] = u.h[j]; }
    __syncthreads();
    v16bf a  = frag16(&AsF[(mt * 32 + lane) * 16]);
    v16bf b0 = load_B_cm(&BsT[(ntb * 16) * BT_LD], BT_LD);
    v16bf b1 = load_B_cm(&BsT[((ntb + 1) * 16) * BT_LD], BT_LD);
    acc0 = wmma_bf16(a, b0, acc0);
    acc1 = wmma_bf16(a, b1, acc1);
    __syncthreads();
  }
  int col = lane & 15, r0 = (lane >> 4) * 8;
#pragma unroll
  for (int i = 0; i < 8; ++i) {
    size_t pos = m0 + mt * 16 + r0 + i;
    R[pos * 2048 + hh * DCC + n0 + ntb * 16 + col]       = (bf16)acc0[i];
    R[pos * 2048 + hh * DCC + n0 + (ntb + 1) * 16 + col] = (bf16)acc1[i];
  }
}

// ---------------------------------------------------------------------------
// Fused attention core: one workgroup per position (4096 WGs, 8 waves).
//   scores = ctx @ R  (A = ctx row-major [vector], B = RlT head-major [vector])
//   masked-bias softmax, one wave per head (wave32 shuffle reductions)
//   T[h,:] = attn_h^T @ ctx  (A = attnT [vector], B = ctx [scalar transpose])
// ---------------------------------------------------------------------------
#define CTX_LD 264
#define RL_LD  264
#define ATT_LD 72
__global__ __launch_bounds__(256)
void k_attn(const float* __restrict__ ctx, const uint8_t* __restrict__ mask,
            const float* __restrict__ bias, const bf16* __restrict__ R,
            bf16* __restrict__ Tg) {
  __shared__ bf16  ctxb[MCTX * CTX_LD];   // row-major [m][c]
  __shared__ bf16  RlT[16 * RL_LD];       // head-major [h][c], rows 8..15 = 0
  __shared__ float Sl[NHEAD * MCTX];
  __shared__ bf16  attnT[16 * ATT_LD];    // [h][m], rows 8..15 = 0
  int pos = blockIdx.x;
  int t = threadIdx.x, wave = t >> 5, lane = t & 31;

  // stream ctx tile -> bf16 LDS (float4 global loads, packed b64 LDS stores)
  const float4* cp = (const float4*)(ctx + (size_t)pos * MCTX * DCC);
#pragma unroll
  for (int rep = 0; rep < 16; ++rep) {
    int i = rep * 1024 + t * 4;
    float4 v = cp[rep * 256 + t];
    uint2 w; w.x = pack2(v.x, v.y); w.y = pack2(v.z, v.w);
    *(uint2*)&ctxb[(i >> 8) * CTX_LD + (i & 255)] = w;
  }
  { // R row for this position -> RlT rows 0..7 (b128 in/out)
    int idx = t * 8;
    v4u val = *(const v4u*)(R + (size_t)pos * 2048 + idx);
    *(v4u*)&RlT[(idx >> 8) * RL_LD + (idx & 255)] = val;
  }
  for (int i = t; i < 8 * RL_LD; i += 256) RlT[8 * RL_LD + i] = (bf16)0.f;
  for (int i = t; i < 8 * ATT_LD; i += 256) attnT[8 * ATT_LD + i] = (bf16)0.f;
  __syncthreads();

  // scores: waves 0..3 own one 16-row m-tile; K = 256 over c.
  if (wave < 4) {
    int mtile = wave;
    v8f acc = {};
#pragma unroll
    for (int kc = 0; kc < 8; ++kc) {
      v16bf a = load_A_rm(&ctxb[mtile * 16 * CTX_LD + kc * 32], CTX_LD);
      v16bf b = load_B_cm(&RlT[kc * 32], RL_LD);
      acc = wmma_bf16(a, b, acc);
    }
    int h = lane & 15, r0 = (lane >> 4) * 8;
    if (h < 8) {
#pragma unroll
      for (int i = 0; i < 8; ++i) Sl[h * MCTX + mtile * 16 + r0 + i] = acc[i];
    }
  }
  __syncthreads();

  // softmax: wave w handles head w; lane covers m=lane, m=lane+32.
  {
    int h = wave;
    int m1 = lane, m2 = lane + 32;
    float s1 = Sl[h * MCTX + m1] + bias[(size_t)pos * MCTX + m1];
    float s2 = Sl[h * MCTX + m2] + bias[(size_t)pos * MCTX + m2];
    if (!mask[(size_t)pos * MCTX + m1]) s1 = -__builtin_inff();
    if (!mask[(size_t)pos * MCTX + m2]) s2 = -__builtin_inff();
    float mx = fmaxf(s1, s2);
#pragma unroll
    for (int off = 16; off; off >>= 1) mx = fmaxf(mx, __shfl_xor(mx, off, 32));
    float e1 = __expf(s1 - mx), e2 = __expf(s2 - mx);
    float sm = e1 + e2;
#pragma unroll
    for (int off = 16; off; off >>= 1) sm += __shfl_xor(sm, off, 32);
    float inv = 1.f / sm;
    attnT[h * ATT_LD + m1] = (bf16)(e1 * inv);
    attnT[h * ATT_LD + m2] = (bf16)(e2 * inv);
  }
  __syncthreads();

  // T = attn^T @ ctx : D tiles 16(h) x 16(c), 16 c-tiles, K = 64 over m.
#pragma unroll
  for (int q = 0; q < 2; ++q) {
    int ct = wave * 2 + q;
    v8f acc = {};
#pragma unroll
    for (int kk = 0; kk < 2; ++kk) {
      int k0 = kk * 32;
      v16bf a = load_A_rm(&attnT[k0], ATT_LD);
      v16bf b = load_B_rm_scalar(&ctxb[k0 * CTX_LD + ct * 16], CTX_LD);
      acc = wmma_bf16(a, b, acc);
    }
    int c = lane & 15, r0 = (lane >> 4) * 8;
#pragma unroll
    for (int i = 0; i < 8; ++i) {
      int h = r0 + i;
      if (h < 8)
        Tg[(size_t)pos * 2048 + h * DCC + ct * 16 + c] = (bf16)acc[i];
    }
  }
}

// ---------------------------------------------------------------------------
// Kernel: out = T @ Wcat + bo.  M=4096 N=512 K=2048, f32 output.
// ---------------------------------------------------------------------------
__global__ __launch_bounds__(256)
void k_gemm_out(const bf16* __restrict__ Tg, const bf16* __restrict__ Wcat,
                const float* __restrict__ bo, float* __restrict__ out) {
  __shared__ bf16 AsF[4 * 32 * 16];
  __shared__ bf16 BsT[64 * BT_LD];
  int m0 = blockIdx.y * 64, n0 = blockIdx.x * 64;
  int t = threadIdx.x, wave = t >> 5, lane = t & 31;
  int mt = wave >> 1, ntb = (wave & 1) * 2;
  v8f acc0 = {}, acc1 = {};
  for (int k0 = 0; k0 < 2048; k0 += 32) {
    { int idx = t * 8, r = idx >> 5, c = idx & 31;
      v4u val = *(const v4u*)(Tg + (size_t)(m0 + r) * 2048 + k0 + c);
      int amt = r >> 4, alane = (r & 15) + ((c >> 3) & 1) * 16;
      *(v4u*)&AsF[(amt * 32 + alane) * 16 + (c >> 4) * 8] = val; }
    { int idx = t * 8, r = idx >> 6, c = idx & 63;
      union { v4u v; bf16 h[8]; } u;
      u.v = *(const v4u*)(Wcat + (size_t)(k0 + r) * DQQ + n0 + c);
#pragma unroll
      for (int j = 0; j < 8; ++j) BsT[(c + j) * BT_LD + r] = u.h[j]; }
    __syncthreads();
    v16bf a  = frag16(&AsF[(mt * 32 + lane) * 16]);
    v16bf b0 = load_B_cm(&BsT[(ntb * 16) * BT_LD], BT_LD);
    v16bf b1 = load_B_cm(&BsT[((ntb + 1) * 16) * BT_LD], BT_LD);
    acc0 = wmma_bf16(a, b0, acc0);
    acc1 = wmma_bf16(a, b1, acc1);
    __syncthreads();
  }
  int col = lane & 15, r0 = (lane >> 4) * 8;
  int c0 = n0 + ntb * 16 + col, c1 = n0 + (ntb + 1) * 16 + col;
  float b0f = bo[c0], b1f = bo[c1];
#pragma unroll
  for (int i = 0; i < 8; ++i) {
    size_t row = m0 + mt * 16 + r0 + i;
    out[row * DQQ + c0] = acc0[i] + b0f;
    out[row * DQQ + c1] = acc1[i] + b1f;
  }
}

// ---------------------------------------------------------------------------
extern "C" void kernel_launch(void* const* d_in, const int* in_sizes, int n_in,
                              void* d_out, int out_size, void* d_ws,
                              size_t ws_size, hipStream_t stream) {
  const float*   x    = (const float*)d_in[0];
  const float*   ctx  = (const float*)d_in[1];
  const uint8_t* mask = (const uint8_t*)d_in[2];  // jax bool = 1 byte/elem
  const float*   bias = (const float*)d_in[3];
  const float*   Wq   = (const float*)d_in[4];
  const float*   Wk   = (const float*)d_in[5];
  const float*   Wv   = (const float*)d_in[6];
  const float*   Wo   = (const float*)d_in[7];
  const float*   bo   = (const float*)d_in[8];
  float*         out  = (float*)d_out;

  char* ws = (char*)d_ws;
  bf16* xb   = (bf16*)(ws + 0);                  //  4 MB  4096x512
  bf16* Wqb  = (bf16*)(ws + 4194304);            //  0.5MB 512x512
  bf16* WkT  = (bf16*)(ws + 4718592);            //  256KB 512x256
  bf16* Wcat = (bf16*)(ws + 4980736);            //  2 MB  2048x512
  bf16* Qs   = (bf16*)(ws + 7077888);            //  4 MB  4096x512
  bf16* R    = (bf16*)(ws + 11272192);           // 16 MB  4096x8x256
  bf16* Tg   = (bf16*)(ws + 28049408);           // 16 MB  4096x2048
  // total 44.8 MB of workspace

  k_f32_to_bf16<<<8192, 256, 0, stream>>>(x, xb, NPOS * DQQ);
  k_f32_to_bf16<<<1024, 256, 0, stream>>>(Wq, Wqb, DQQ * DQQ);
  k_build_wkt<<<512, 256, 0, stream>>>(Wk, WkT);
  k_build_wcat<<<4096, 256, 0, stream>>>(Wv, Wo, Wcat);

  k_gemm_q<<<dim3(8, 64), 256, 0, stream>>>(xb, Wqb, Qs);
  k_gemm_r<<<dim3(4, 64, 8), 256, 0, stream>>>(Qs, WkT, R);
  k_attn<<<NPOS, 256, 0, stream>>>(ctx, mask, bias, R, Tg);
  k_gemm_out<<<dim3(8, 64), 256, 0, stream>>>(Tg, Wcat, bo, out);
}
